// TransformerBlock_76115410419716
// MI455X (gfx1250) — matrix operand order repaired and checked by reading, not compiled, not run on hardware
//
#include <hip/hip_runtime.h>
#include <hip/hip_bf16.h>
#include <cstdint>
#include <cstddef>

// ---------------------------------------------------------------------------
// Transformer block for MI455X (gfx1250, wave32, WMMA).
// GEMMs: v_wmma_f32_16x16x32_bf16, B tiles staged in LDS via
// GLOBAL_LOAD_ASYNC_TO_LDS_B128 (ASYNCcnt double-buffer), A frags from global.
// ---------------------------------------------------------------------------

#define EMBED 768
#define HEADS 12
#define HID   (EMBED * HEADS)   // 9216
#define BATCH 4
#define SEQ   1024
#define TOK   (BATCH * SEQ)     // 4096

typedef __attribute__((ext_vector_type(16))) __bf16 v16bf;
typedef __attribute__((ext_vector_type(8)))  float  v8f;
typedef __attribute__((ext_vector_type(4)))  int    v4i;

#if defined(__HIP_DEVICE_COMPILE__) && \
    __has_builtin(__builtin_amdgcn_global_load_async_to_lds_b128) && \
    __has_builtin(__builtin_amdgcn_s_wait_asynccnt)
#define HAVE_ASYNC_LDS 1
#else
#define HAVE_ASYNC_LDS 0
#endif

#define AS1 __attribute__((address_space(1)))
#define AS3 __attribute__((address_space(3)))

// fp32 -> bf16, round-to-nearest-even
__device__ __forceinline__ unsigned short f2bf(float f) {
  unsigned u = __float_as_uint(f);
  u += 0x7fffu + ((u >> 16) & 1u);
  return (unsigned short)(u >> 16);
}

union FragU { uint4 u[2]; v16bf v; };

// A fragment: 16x32 bf16, row-major source (lda in elements).
// lanes 0-15 -> rows 0-15 with K {0..7,16..23}; lanes 16-31 -> K {8..15,24..31}.
__device__ __forceinline__ v16bf load_a_frag(const unsigned short* p0, int ld, int lane) {
  int r  = lane & 15;
  int kh = (lane >> 4) << 3;            // 0 or 8
  const unsigned short* p = p0 + r * ld + kh;
  FragU f;
  f.u[0] = *reinterpret_cast<const uint4*>(p);
  f.u[1] = *reinterpret_cast<const uint4*>(p + 16);
  return f.v;
}

// ---------------- LDS-staged B tile -----------------------------------------
// B tile in LDS: 64 rows (output columns n) x 32 (K-step) bf16,
// padded row stride 40 elements (80 B) -> conflict-free ds_load_b128.
#define BROWS 64
#define BPAD  40
#define BTILE (BROWS * BPAD)   // elements per buffer

// B fragment from LDS: lanes 0-15 -> cols 0-15, K {0..7,16..23}; lanes 16-31 -> K {8..15,24..31}.
__device__ __forceinline__ v16bf lds_b_frag(const unsigned short* sbase, int j, int lane) {
  int c  = lane & 15;
  int kh = (lane >> 4) << 3;            // 0 or 8
  const unsigned short* p = sbase + (j * 16 + c) * BPAD + kh;
  FragU f;
  f.u[0] = *reinterpret_cast<const uint4*>(p);
  f.u[1] = *reinterpret_cast<const uint4*>(p + 16);
  return f.v;
}

// Cooperative fill of one 64x32 bf16 tile: 256 threads x one b128 chunk each.
// Async path: GLOBAL_LOAD_ASYNC_TO_LDS_B128 (ASYNCcnt); fallback: ld+ds_store.
__device__ __forceinline__ void stage_issue(const unsigned short* src, int ldbt,
                                            unsigned short* dst, int tid) {
  int row = tid >> 2;          // 0..63
  int col = (tid & 3) * 8;     // 0,8,16,24 elements (16B chunks)
  const unsigned short* g = src + (size_t)row * ldbt + col;
  unsigned short* l = dst + row * BPAD + col;
#if HAVE_ASYNC_LDS
  __builtin_amdgcn_global_load_async_to_lds_b128(
      (AS1 v4i*)g, (AS3 v4i*)l, 0, 0);
#else
  *reinterpret_cast<uint4*>(l) = *reinterpret_cast<const uint4*>(g);
#endif
}

__device__ __forceinline__ v8f vzero() {
  v8f z = {0.f, 0.f, 0.f, 0.f, 0.f, 0.f, 0.f, 0.f};
  return z;
}

// Wave computes a 16(M) x 64(N) tile; block = 8 waves stacked in M sharing the
// LDS-staged B tile.  Double-buffered: tile k+32 is in flight while tile k is
// consumed by 4 WMMAs per wave.  All four B fragments are hoisted into live
// registers before the WMMAs so the ds_load_b128s pipeline in one clause.
__device__ __forceinline__ void wave_gemm4_lds(const unsigned short* A, int lda,
                                               const unsigned short* BT, int ldbt,
                                               int K, unsigned short* sB,
                                               int tid, v8f acc[4]) {
  int lane = tid & 31;
  stage_issue(BT, ldbt, sB, tid);                       // prime buffer 0
  for (int k = 0; k < K; k += 32) {
    unsigned short* cur = sB + ((k >> 5) & 1) * BTILE;
    unsigned short* nxt = sB + (((k >> 5) + 1) & 1) * BTILE;
    bool more = (k + 32) < K;
    if (more) stage_issue(BT + k + 32, ldbt, nxt, tid); // overlap next tile
#if HAVE_ASYNC_LDS
    if (more) __builtin_amdgcn_s_wait_asynccnt(1);      // current tile done
    else      __builtin_amdgcn_s_wait_asynccnt(0);
#endif
    __syncthreads();                                    // tile visible to all
    v16bf a = load_a_frag(A + k, lda, lane);
    if (more) __builtin_prefetch(A + k + 32, 0, 0);     // global_prefetch_b8
    // Hoist all B fragments (distinct registers) so ds loads clause together.
    v16bf b0 = lds_b_frag(cur, 0, lane);
    v16bf b1 = lds_b_frag(cur, 1, lane);
    v16bf b2 = lds_b_frag(cur, 2, lane);
    v16bf b3 = lds_b_frag(cur, 3, lane);
    acc[0] = __builtin_amdgcn_wmma_f32_16x16x32_bf16(
        false, a, false, b0, (short)0, acc[0], false, false);
    acc[1] = __builtin_amdgcn_wmma_f32_16x16x32_bf16(
        false, a, false, b1, (short)0, acc[1], false, false);
    acc[2] = __builtin_amdgcn_wmma_f32_16x16x32_bf16(
        false, a, false, b2, (short)0, acc[2], false, false);
    acc[3] = __builtin_amdgcn_wmma_f32_16x16x32_bf16(
        false, a, false, b3, (short)0, acc[3], false, false);
    __syncthreads();                                    // done reading cur
  }
}

// ---------------------------------------------------------------------------
// Weight transpose + fp32->bf16:  W[K][N] (f32) -> WT[N][K] (bf16)
// ---------------------------------------------------------------------------
__global__ __launch_bounds__(256)
void k_transpose_bf16(const float* __restrict__ W, unsigned short* __restrict__ WT,
                      int K, int N) {
  __shared__ float tile[32][33];
  int bx = blockIdx.x * 32;   // N
  int by = blockIdx.y * 32;   // K
  int tx = threadIdx.x;       // 0..31
  int ty = threadIdx.y;       // 0..7
  for (int i = ty; i < 32; i += 8)
    tile[i][tx] = W[(size_t)(by + i) * N + (bx + tx)];
  __syncthreads();
  for (int i = ty; i < 32; i += 8)
    WT[(size_t)(bx + i) * K + (by + tx)] = f2bf(tile[tx][i]);
}

// ---------------------------------------------------------------------------
// LayerNorm row (EMBED=768, 256 threads, 3 elems/thread) -> bf16
// ---------------------------------------------------------------------------
__global__ __launch_bounds__(256)
void k_layernorm(const float* __restrict__ x, const float* __restrict__ g,
                 const float* __restrict__ bt, unsigned short* __restrict__ out) {
  __shared__ float sh[256];
  int row = blockIdx.x, tid = threadIdx.x;
  const float* xr = x + (size_t)row * EMBED;
  float v0 = xr[tid], v1 = xr[tid + 256], v2 = xr[tid + 512];
  sh[tid] = v0 + v1 + v2;
  __syncthreads();
  for (int o = 128; o > 0; o >>= 1) { if (tid < o) sh[tid] += sh[tid + o]; __syncthreads(); }
  float mu = sh[0] * (1.0f / EMBED);
  __syncthreads();
  float d0 = v0 - mu, d1 = v1 - mu, d2 = v2 - mu;
  sh[tid] = d0 * d0 + d1 * d1 + d2 * d2;
  __syncthreads();
  for (int o = 128; o > 0; o >>= 1) { if (tid < o) sh[tid] += sh[tid + o]; __syncthreads(); }
  float rstd = rsqrtf(sh[0] * (1.0f / EMBED) + 1e-5f);
  unsigned short* orow = out + (size_t)row * EMBED;
  orow[tid]       = f2bf(d0 * rstd * g[tid]       + bt[tid]);
  orow[tid + 256] = f2bf(d1 * rstd * g[tid + 256] + bt[tid + 256]);
  orow[tid + 512] = f2bf(d2 * rstd * g[tid + 512] + bt[tid + 512]);
}

// ---------------------------------------------------------------------------
// Q/K projection: NX[4096x768] @ W^T + bias -> bf16 [4096x9216]
// grid (HID/64, TOK/128), block 256 (8 waves in M)
// ---------------------------------------------------------------------------
__global__ __launch_bounds__(256)
void k_gemm_qk(const unsigned short* __restrict__ NX, const unsigned short* __restrict__ WT,
               const float* __restrict__ bias, unsigned short* __restrict__ out) {
  __shared__ unsigned short sB[2 * BTILE];
  int tid = threadIdx.x, lane = tid & 31, wave = tid >> 5;
  int m0 = blockIdx.y * 128 + wave * 16;
  int n0 = blockIdx.x * 64;
  v8f acc[4]; for (int j = 0; j < 4; ++j) acc[j] = vzero();
  wave_gemm4_lds(NX + (size_t)m0 * EMBED, EMBED, WT + (size_t)n0 * EMBED, EMBED,
                 EMBED, sB, tid, acc);
  int r  = m0 + (lane >> 4) * 8;
  int c0 = n0 + (lane & 15);
#pragma unroll
  for (int j = 0; j < 4; ++j) {
    int n = c0 + j * 16;
    float bn = bias[n];
#pragma unroll
    for (int i = 0; i < 8; ++i)
      out[(size_t)(r + i) * HID + n] = f2bf(acc[j][i] + bn);
  }
}

// V projection with per-head transposed store: VT[b][h][d][s]
__global__ __launch_bounds__(256)
void k_gemm_v(const unsigned short* __restrict__ NX, const unsigned short* __restrict__ WT,
              const float* __restrict__ bias, unsigned short* __restrict__ VT) {
  __shared__ unsigned short sB[2 * BTILE];
  int tid = threadIdx.x, lane = tid & 31, wave = tid >> 5;
  int m0 = blockIdx.y * 128 + wave * 16;
  int n0 = blockIdx.x * 64;
  v8f acc[4]; for (int j = 0; j < 4; ++j) acc[j] = vzero();
  wave_gemm4_lds(NX + (size_t)m0 * EMBED, EMBED, WT + (size_t)n0 * EMBED, EMBED,
                 EMBED, sB, tid, acc);
  int r  = m0 + (lane >> 4) * 8;
  int c0 = n0 + (lane & 15);
#pragma unroll
  for (int j = 0; j < 4; ++j) {
    int n = c0 + j * 16;
    int h = n / EMBED, d = n % EMBED;
    float bn = bias[n];
#pragma unroll
    for (int i = 0; i < 8; ++i) {
      int m = r + i;
      int b = m >> 10, s = m & (SEQ - 1);
      VT[((size_t)(b * HEADS + h) * EMBED + d) * SEQ + s] = f2bf(acc[j][i] + bn);
    }
  }
}

// ---------------------------------------------------------------------------
// Scores: per (b,h): Q_bh[1024x768] @ K_bh^T -> f32 * 1/sqrt(768)
// grid (SEQ/64, SEQ/128, 48)
// ---------------------------------------------------------------------------
__global__ __launch_bounds__(256)
void k_gemm_scores(const unsigned short* __restrict__ Q, const unsigned short* __restrict__ Kc,
                   float* __restrict__ scores) {
  __shared__ unsigned short sB[2 * BTILE];
  int z = blockIdx.z;                 // b*HEADS + h
  int b = z / HEADS, h = z % HEADS;
  const unsigned short* A  = Q  + (size_t)b * SEQ * HID + h * EMBED;
  const unsigned short* BT = Kc + (size_t)b * SEQ * HID + h * EMBED;
  float* out = scores + (size_t)z * SEQ * SEQ;
  int tid = threadIdx.x, lane = tid & 31, wave = tid >> 5;
  int m0 = blockIdx.y * 128 + wave * 16;
  int n0 = blockIdx.x * 64;
  v8f acc[4]; for (int j = 0; j < 4; ++j) acc[j] = vzero();
  wave_gemm4_lds(A + (size_t)m0 * HID, HID, BT + (size_t)n0 * HID, HID,
                 EMBED, sB, tid, acc);
  const float scale = 0.03608439182435161f;   // 1/sqrt(768)
  int r  = m0 + (lane >> 4) * 8;
  int c0 = n0 + (lane & 15);
#pragma unroll
  for (int j = 0; j < 4; ++j)
#pragma unroll
    for (int i = 0; i < 8; ++i)
      out[(size_t)(r + i) * SEQ + c0 + j * 16] = acc[j][i] * scale;
}

// ---------------------------------------------------------------------------
// Softmax over the QUERY axis (axis=-2). Thread owns one k column. -> bf16
// grid (SEQ/256, 48), block 256
// ---------------------------------------------------------------------------
__global__ __launch_bounds__(256)
void k_softmax_q(const float* __restrict__ scores, unsigned short* __restrict__ attn) {
  int z = blockIdx.y;
  int k = blockIdx.x * 256 + threadIdx.x;
  const float* s = scores + (size_t)z * SEQ * SEQ + k;
  unsigned short* a = attn + (size_t)z * SEQ * SEQ + k;
  float mx = -3.4e38f, sum = 0.f;
  for (int q = 0; q < SEQ; ++q) {
    float v  = s[(size_t)q * SEQ];
    float nm = fmaxf(mx, v);
    sum = sum * __expf(mx - nm) + __expf(v - nm);
    mx = nm;
  }
  float inv = 1.0f / sum;
  for (int q = 0; q < SEQ; ++q)
    a[(size_t)q * SEQ] = f2bf(__expf(s[(size_t)q * SEQ] - mx) * inv);
}

// ---------------------------------------------------------------------------
// Context: per (b,h): attn[1024x1024] @ V_bh -> ctx bf16 [tok][HID]
// V pre-transposed VT[b][h][d][s].  grid (EMBED/64, SEQ/128, 48)
// ---------------------------------------------------------------------------
__global__ __launch_bounds__(256)
void k_gemm_ctx(const unsigned short* __restrict__ ATT, const unsigned short* __restrict__ VT,
                unsigned short* __restrict__ ctx) {
  __shared__ unsigned short sB[2 * BTILE];
  int z = blockIdx.z;
  int b = z / HEADS, h = z % HEADS;
  const unsigned short* A  = ATT + (size_t)z * SEQ * SEQ;     // [q][k], ld SEQ
  const unsigned short* BT = VT  + (size_t)z * EMBED * SEQ;   // [d][s], ld SEQ
  int tid = threadIdx.x, lane = tid & 31, wave = tid >> 5;
  int m0 = blockIdx.y * 128 + wave * 16;   // q
  int n0 = blockIdx.x * 64;                // d
  v8f acc[4]; for (int j = 0; j < 4; ++j) acc[j] = vzero();
  wave_gemm4_lds(A + (size_t)m0 * SEQ, SEQ, BT + (size_t)n0 * SEQ, SEQ,
                 SEQ, sB, tid, acc);
  int r  = m0 + (lane >> 4) * 8;
  int c0 = n0 + (lane & 15);
#pragma unroll
  for (int j = 0; j < 4; ++j)
#pragma unroll
    for (int i = 0; i < 8; ++i)
      ctx[((size_t)(b * SEQ + r + i)) * HID + h * EMBED + c0 + j * 16] =
          f2bf(acc[j][i]);
}

// ---------------------------------------------------------------------------
// Output projection + residual: ctx @ wo^T + bo + x -> xr (f32)
// grid (EMBED/64, TOK/128)
// ---------------------------------------------------------------------------
__global__ __launch_bounds__(256)
void k_gemm_out(const unsigned short* __restrict__ CTX, const unsigned short* __restrict__ WOT,
                const float* __restrict__ bo, const float* __restrict__ x,
                float* __restrict__ xr) {
  __shared__ unsigned short sB[2 * BTILE];
  int tid = threadIdx.x, lane = tid & 31, wave = tid >> 5;
  int m0 = blockIdx.y * 128 + wave * 16;
  int n0 = blockIdx.x * 64;
  v8f acc[4]; for (int j = 0; j < 4; ++j) acc[j] = vzero();
  wave_gemm4_lds(CTX + (size_t)m0 * HID, HID, WOT + (size_t)n0 * HID, HID,
                 HID, sB, tid, acc);
  int r  = m0 + (lane >> 4) * 8;
  int c0 = n0 + (lane & 15);
#pragma unroll
  for (int j = 0; j < 4; ++j) {
    int n = c0 + j * 16;
    float bn = bo[n];
#pragma unroll
    for (int i = 0; i < 8; ++i) {
      size_t idx = (size_t)(r + i) * EMBED + n;
      xr[idx] = acc[j][i] + bn + x[idx];
    }
  }
}

// MLP fc1 + exact GELU -> bf16.  grid (EMBED/64, TOK/128)
__global__ __launch_bounds__(256)
void k_gemm_mlp1(const unsigned short* __restrict__ H2, const unsigned short* __restrict__ W1T,
                 const float* __restrict__ b1, unsigned short* __restrict__ M1) {
  __shared__ unsigned short sB[2 * BTILE];
  int tid = threadIdx.x, lane = tid & 31, wave = tid >> 5;
  int m0 = blockIdx.y * 128 + wave * 16;
  int n0 = blockIdx.x * 64;
  v8f acc[4]; for (int j = 0; j < 4; ++j) acc[j] = vzero();
  wave_gemm4_lds(H2 + (size_t)m0 * EMBED, EMBED, W1T + (size_t)n0 * EMBED, EMBED,
                 EMBED, sB, tid, acc);
  int r  = m0 + (lane >> 4) * 8;
  int c0 = n0 + (lane & 15);
#pragma unroll
  for (int j = 0; j < 4; ++j) {
    int n = c0 + j * 16;
    float bn = b1[n];
#pragma unroll
    for (int i = 0; i < 8; ++i) {
      float t = acc[j][i] + bn;
      float g = 0.5f * t * (1.0f + erff(t * 0.70710678118654752f));
      M1[(size_t)(r + i) * EMBED + n] = f2bf(g);
    }
  }
}

// MLP fc2 + residual -> final f32 output.  grid (EMBED/64, TOK/128)
__global__ __launch_bounds__(256)
void k_gemm_mlp2(const unsigned short* __restrict__ M1, const unsigned short* __restrict__ W2T,
                 const float* __restrict__ b2, const float* __restrict__ xr,
                 float* __restrict__ out) {
  __shared__ unsigned short sB[2 * BTILE];
  int tid = threadIdx.x, lane = tid & 31, wave = tid >> 5;
  int m0 = blockIdx.y * 128 + wave * 16;
  int n0 = blockIdx.x * 64;
  v8f acc[4]; for (int j = 0; j < 4; ++j) acc[j] = vzero();
  wave_gemm4_lds(M1 + (size_t)m0 * EMBED, EMBED, W2T + (size_t)n0 * EMBED, EMBED,
                 EMBED, sB, tid, acc);
  int r  = m0 + (lane >> 4) * 8;
  int c0 = n0 + (lane & 15);
#pragma unroll
  for (int j = 0; j < 4; ++j) {
    int n = c0 + j * 16;
    float bn = b2[n];
#pragma unroll
    for (int i = 0; i < 8; ++i) {
      size_t idx = (size_t)(r + i) * EMBED + n;
      out[idx] = acc[j][i] + bn + xr[idx];
    }
  }
}

// ---------------------------------------------------------------------------
extern "C" void kernel_launch(void* const* d_in, const int* in_sizes, int n_in,
                              void* d_out, int out_size, void* d_ws, size_t ws_size,
                              hipStream_t stream) {
  (void)in_sizes; (void)n_in; (void)out_size; (void)ws_size;
  const float* x    = (const float*)d_in[0];
  const float* ln1g = (const float*)d_in[1];
  const float* ln1b = (const float*)d_in[2];
  const float* wq   = (const float*)d_in[3];
  const float* bq   = (const float*)d_in[4];
  const float* wk   = (const float*)d_in[5];
  const float* bk   = (const float*)d_in[6];
  const float* wv   = (const float*)d_in[7];
  const float* bv   = (const float*)d_in[8];
  const float* wo   = (const float*)d_in[9];
  const float* bo   = (const float*)d_in[10];
  const float* ln2g = (const float*)d_in[11];
  const float* ln2b = (const float*)d_in[12];
  const float* w1   = (const float*)d_in[13];
  const float* b1   = (const float*)d_in[14];
  const float* w2   = (const float*)d_in[15];
  const float* b2   = (const float*)d_in[16];

  // -------- workspace carve-up (256B aligned) --------
  char* ws = (char*)d_ws;
  size_t off = 0;
  auto carve = [&](size_t bytes) -> void* {
    void* p = ws + off;
    off += (bytes + 255) & ~(size_t)255;
    return p;
  };
  unsigned short* WQT = (unsigned short*)carve((size_t)HID * EMBED * 2);   // [9216][768]
  unsigned short* WKT = (unsigned short*)carve((size_t)HID * EMBED * 2);
  unsigned short* WVT = (unsigned short*)carve((size_t)HID * EMBED * 2);
  unsigned short* WOT = (unsigned short*)carve((size_t)EMBED * HID * 2);   // [768][9216]
  unsigned short* W1T = (unsigned short*)carve((size_t)EMBED * EMBED * 2);
  unsigned short* W2T = (unsigned short*)carve((size_t)EMBED * EMBED * 2);
  unsigned short* NX  = (unsigned short*)carve((size_t)TOK * EMBED * 2);
  unsigned short* Qb  = (unsigned short*)carve((size_t)TOK * HID * 2);     // 75.5 MB
  unsigned short* Kb  = (unsigned short*)carve((size_t)TOK * HID * 2);     // 75.5 MB
  unsigned short* VT  = (unsigned short*)carve((size_t)TOK * HID * 2);     // [b][h][d][s]
  float*          SC  = (float*)carve((size_t)BATCH * HEADS * SEQ * SEQ * 4); // 201 MB
  float*          XR  = (float*)carve((size_t)TOK * EMBED * 4);
  unsigned short* H2  = (unsigned short*)carve((size_t)TOK * EMBED * 2);
  unsigned short* M1  = (unsigned short*)carve((size_t)TOK * EMBED * 2);
  // Aliases over dead buffers (Q,K dead after scores; scores dead after softmax):
  unsigned short* ATT = Qb;                   // 100.7 MB into Q∪K (151 MB contiguous)
  unsigned short* CTX = (unsigned short*)SC;  // 75.5 MB into scores region

  dim3 tb(32, 8);
  // weight transpose+convert
  k_transpose_bf16<<<dim3(HID / 32, EMBED / 32), tb, 0, stream>>>(wq, WQT, EMBED, HID);
  k_transpose_bf16<<<dim3(HID / 32, EMBED / 32), tb, 0, stream>>>(wk, WKT, EMBED, HID);
  k_transpose_bf16<<<dim3(HID / 32, EMBED / 32), tb, 0, stream>>>(wv, WVT, EMBED, HID);
  k_transpose_bf16<<<dim3(EMBED / 32, HID / 32), tb, 0, stream>>>(wo, WOT, HID, EMBED);
  k_transpose_bf16<<<dim3(EMBED / 32, EMBED / 32), tb, 0, stream>>>(w1, W1T, EMBED, EMBED);
  k_transpose_bf16<<<dim3(EMBED / 32, EMBED / 32), tb, 0, stream>>>(w2, W2T, EMBED, EMBED);

  // LN1 -> bf16
  k_layernorm<<<TOK, 256, 0, stream>>>(x, ln1g, ln1b, NX);

  // QKV projections (V stored transposed per head)
  dim3 gProj(HID / 64, TOK / 128);
  k_gemm_qk<<<gProj, 256, 0, stream>>>(NX, WQT, bq, Qb);
  k_gemm_qk<<<gProj, 256, 0, stream>>>(NX, WKT, bk, Kb);
  k_gemm_v <<<gProj, 256, 0, stream>>>(NX, WVT, bv, VT);

  // scores, softmax over query axis, context
  k_gemm_scores<<<dim3(SEQ / 64, SEQ / 128, BATCH * HEADS), 256, 0, stream>>>(Qb, Kb, SC);
  k_softmax_q  <<<dim3(SEQ / 256, BATCH * HEADS), 256, 0, stream>>>(SC, ATT);
  k_gemm_ctx   <<<dim3(EMBED / 64, SEQ / 128, BATCH * HEADS), 256, 0, stream>>>(ATT, VT, CTX);

  // out-projection + residual, LN2, MLP
  dim3 gThin(EMBED / 64, TOK / 128);
  k_gemm_out <<<gThin, 256, 0, stream>>>(CTX, WOT, bo, x, XR);
  k_layernorm<<<TOK, 256, 0, stream>>>(XR, ln2g, ln2b, H2);
  k_gemm_mlp1<<<gThin, 256, 0, stream>>>(H2, W1T, b1, M1);
  k_gemm_mlp2<<<gThin, 256, 0, stream>>>(M1, W2T, b2, XR, (float*)d_out);
}
